// ImageTextGeneratorModel_11244224381634
// MI455X (gfx1250) — matrix operand order repaired and checked by reading, not compiled, not run on hardware
//
#include <hip/hip_runtime.h>
#include <hip/hip_bf16.h>

// ---------------------------------------------------------------------------
// Model dims
// ---------------------------------------------------------------------------
#define Vv    32000
#define INw   300
#define INP   320          // IN padded to multiple of 64 (K-slab)
#define Hh    1024
#define Ff    1024
#define Ee    512
#define IMG   4096
#define Tt    32
#define Bb    128
#define TB    4096         // T*B
#define G4    4096         // 4*F == 4*H

// LDS staging geometry (per stage): A tile 64x64 bf16, B tile 128x64 bf16,
// rows padded to 72 elements (144B = 36 banks; 36*r mod 64 distinct, r=0..15)
#define KSLAB       64
#define LROW        72
#define A_ELEMS     (64 * LROW)              // 4608 elems,  9216 B
#define STAGE_ELEMS (A_ELEMS + 128 * LROW)   // 13824 elems, 27648 B
#define STAGE_BYTES (STAGE_ELEMS * 2)

typedef __bf16 bf16;
typedef __attribute__((ext_vector_type(16))) __bf16 v16bf;
typedef __attribute__((ext_vector_type(8)))  float  v8f;
typedef unsigned int u32x4 __attribute__((ext_vector_type(4)));

union Frag { v16bf v; u32x4 q[2]; };

__device__ __forceinline__ bf16 f2b(float f) {
    unsigned u = __builtin_bit_cast(unsigned, f);
    unsigned r = u + 0x7FFFu + ((u >> 16) & 1u);       // round-to-nearest-even
    return __builtin_bit_cast(bf16, (unsigned short)(r >> 16));
}

// CDNA5 async global->LDS copy (ASYNCcnt path); each lane moves 16 bytes.
__device__ __forceinline__ void async_g2l_b128(unsigned lds_byte, const void* gaddr) {
    asm volatile("global_load_async_to_lds_b128 %0, %1, off"
                 :: "v"(lds_byte), "v"(gaddr) : "memory");
}
__device__ __forceinline__ void wait_async0() {
    asm volatile("s_wait_asynccnt 0x0" ::: "memory");
}

// ---------------------------------------------------------------------------
// Flexible bf16 WMMA GEMM:  C[M,N] = A[M,K] * Bw[N,K]^T   (+bias, *mul, +add)
//   - block: 256 threads = 8 waves; wave tile 16x64; block tile 64x128
//   - A/B K=64 slabs staged in LDS via async b128 loads, double-buffered
//   - per K=32 step: load A + all 4 B fragments (simultaneously live), then
//     issue 4 WMMAs -> partial dscnt waits instead of full drains
//   - grid.z batches gates (pointer offsets a_z / b_z / *_z per z)
//   - outputs: f32 (outF) and/or bf16 (outB); all epilogue ptrs optional
// ---------------------------------------------------------------------------
__global__ __launch_bounds__(256) void k_gemm(
    const bf16* __restrict__ A, int lda, long long a_z,
    const bf16* __restrict__ Bw, int ldb, long long b_z,
    int N, int K,
    float* outF, int ldoF, long long oF_z,
    bf16* outB, int ldoB, long long oB_z,
    const float* bias, long long bias_z,
    const float* mul, int ldmul,
    const float* add, int ldadd, long long add_z)
{
    __shared__ bf16 smem[2 * STAGE_ELEMS];

    const int z = blockIdx.z;
    A  += (long long)z * a_z;
    Bw += (long long)z * b_z;
    if (outF) outF += (long long)z * oF_z;
    if (outB) outB += (long long)z * oB_z;
    if (bias) bias += (long long)z * bias_z;
    if (add)  add  += (long long)z * add_z;

    const int tid  = threadIdx.x;
    const int lane = tid & 31;
    const int w    = tid >> 5;
    const int wm   = w >> 1, wn = w & 1;
    const int m0   = blockIdx.x * 64  + wm * 16;
    const int n0   = blockIdx.y * 128 + wn * 64;
    const int lr   = lane & 15, lh = lane >> 4;

    // async-copy source pointers (per thread per stage: 2 A chunks, 4 B chunks)
    const bf16* gA = A  + (long long)(blockIdx.x * 64  + (tid >> 2)) * lda + (tid & 3) * 8;
    const bf16* gB = Bw + (long long)(blockIdx.y * 128 + (tid >> 1)) * ldb + (tid & 1) * 8;
    const unsigned sb0  = (unsigned)(unsigned long long)&smem[0];
    const unsigned aoff = (unsigned)((tid >> 2) * (LROW * 2) + (tid & 3) * 16);
    const unsigned boff = (unsigned)(A_ELEMS * 2 + (tid >> 1) * (LROW * 2) + (tid & 1) * 16);

    v8f acc[4] = { v8f{}, v8f{}, v8f{}, v8f{} };

    const int nk = K / KSLAB;
    // prologue: stage 0
    {
        #pragma unroll
        for (int i = 0; i < 2; ++i)
            async_g2l_b128(sb0 + aoff + i * 64, gA + i * 32);
        #pragma unroll
        for (int i = 0; i < 4; ++i)
            async_g2l_b128(sb0 + boff + i * 32, gB + i * 16);
    }

    for (int kk = 0; kk < nk; ++kk) {
        wait_async0();            // our share of stage kk landed in LDS
        __syncthreads();          // everyone's share landed
        if (kk + 1 < nk) {        // prefetch stage kk+1 into the other buffer
            const unsigned sb = sb0 + (unsigned)(((kk + 1) & 1) * STAGE_BYTES);
            const int k1 = (kk + 1) * KSLAB;
            #pragma unroll
            for (int i = 0; i < 2; ++i)
                async_g2l_b128(sb + aoff + i * 64, gA + k1 + i * 32);
            #pragma unroll
            for (int i = 0; i < 4; ++i)
                async_g2l_b128(sb + boff + i * 32, gB + k1 + i * 16);
        }
        const bf16* sAb = smem + (kk & 1) * STAGE_ELEMS + (wm * 16 + lr) * LROW + lh * 16;
        const bf16* sBb = smem + (kk & 1) * STAGE_ELEMS + A_ELEMS
                               + (wn * 64 + lr) * LROW + lh * 16;
        #pragma unroll
        for (int ks = 0; ks < 2; ++ks) {   // two K=32 steps per staged slab
            // phase 1: load A + all four B fragments (all live at once)
            const bf16* sA = sAb + ks * 32;
            Frag a, b[4];
            a.q[0] = *(const u32x4*)(sA);
            a.q[1] = *(const u32x4*)(sA + 8);
            #pragma unroll
            for (int j = 0; j < 4; ++j) {
                const bf16* bp = sBb + ks * 32 + j * 16 * LROW;
                b[j].q[0] = *(const u32x4*)(bp);
                b[j].q[1] = *(const u32x4*)(bp + 8);
            }
            // phase 2: issue the four WMMAs
            #pragma unroll
            for (int j = 0; j < 4; ++j) {
                acc[j] = __builtin_amdgcn_wmma_f32_16x16x32_bf16(
                    false, a.v, false, b[j].v, (short)0, acc[j], false, false);
            }
        }
        __syncthreads();          // stage kk buffer free for reuse
    }

    // Epilogue. C/D layout: VGPR r, lanes 0-15 -> M=r, lanes 16-31 -> M=8+r; N=lr
    const int gmBase = m0 + lh * 8;
    #pragma unroll
    for (int j = 0; j < 4; ++j) {
        const int gn = n0 + j * 16 + lr;
        if (gn < N) {
            const float bv = bias ? bias[gn] : 0.0f;
            #pragma unroll
            for (int r = 0; r < 8; ++r) {
                const int gm = gmBase + r;
                float v = acc[j][r] + bv;
                if (mul) v *= mul[(long long)(gm & (Bb - 1)) * ldmul + gn];
                if (add) v += add[(long long)gm * ldadd + gn];
                if (outF) outF[(long long)gm * ldoF + gn] = v;
                if (outB) outB[(long long)gm * ldoB + gn] = f2b(v);
            }
        }
    }
}

// ---------------------------------------------------------------------------
// Helpers: conversion, padding, x build, gates, zero, tail copy
// ---------------------------------------------------------------------------
__global__ void k_cvt(bf16* dst, const float* src, long long n) {
    long long i = (long long)blockIdx.x * blockDim.x + threadIdx.x;
    if (i < n) dst[i] = f2b(src[i]);
}

__global__ void k_cvt_pad(bf16* dst, const float* src, int drows, int dcols,
                          int srows, int scols) {
    long long i = (long long)blockIdx.x * blockDim.x + threadIdx.x;
    long long total = (long long)drows * dcols;
    if (i >= total) return;
    int r = (int)(i / dcols), c = (int)(i % dcols);
    float v = (r < srows && c < scols) ? src[(long long)r * scols + c] : 0.0f;
    dst[i] = f2b(v);
}

__global__ void k_zero_b(bf16* p, long long n) {
    long long i = (long long)blockIdx.x * blockDim.x + threadIdx.x;
    if (i < n) p[i] = (bf16)0.0f;
}
__global__ void k_zero_f(float* p, long long n) {
    long long i = (long long)blockIdx.x * blockDim.x + threadIdx.x;
    if (i < n) p[i] = 0.0f;
}

// x_bf[T*B, INP]: rows t>=1 gather embedding (bf16, K-padded); t==0 row padding
// only (img GEMM fills cols < INw afterwards).
__global__ void k_build_x(bf16* x, const float* emb, const int* seqs) {
    int row = blockIdx.x;            // 0..TB-1
    int col = threadIdx.x;           // 0..INP-1
    int t = row >> 7, b = row & (Bb - 1);
    bf16 v = (bf16)0.0f;
    if (col < INw && t >= 1) {
        int tok = seqs[t * Bb + b];
        v = f2b(emb[(long long)tok * INw + col]);
    }
    if (t >= 1 || col >= INw) x[(long long)row * INP + col] = v;
}

__global__ void k_gates(const float* __restrict__ zb, float* __restrict__ c,
                        float* __restrict__ hF, bf16* __restrict__ hB,
                        bf16* __restrict__ hs, int t) {
    int idx = blockIdx.x * blockDim.x + threadIdx.x;   // 0 .. B*H-1
    int b = idx >> 10, hcol = idx & (Hh - 1);
    const float* zr = zb + (long long)b * G4;
    float zi = zr[hcol], zf = zr[Hh + hcol], zo = zr[2 * Hh + hcol], zg = zr[3 * Hh + hcol];
    float ig = 1.0f / (1.0f + expf(-zi));
    float fg = 1.0f / (1.0f + expf(-zf));
    float og = 1.0f / (1.0f + expf(-zo));
    float gg = tanhf(zg);
    float cn = fg * c[idx] + ig * gg;
    float hn = og * tanhf(cn);
    c[idx]  = cn;
    hF[idx] = hn;
    bf16 hb = f2b(hn);
    hB[idx] = hb;
    hs[(long long)t * (Bb * Hh) + idx] = hb;
}

__global__ void k_tail(float* out, const float* hF, const float* cF) {
    int i = blockIdx.x * blockDim.x + threadIdx.x;     // 0 .. B*H-1
    out[i] = hF[i];
    out[Bb * Hh + i] = cF[i];
}

// ---------------------------------------------------------------------------
// Launch
// ---------------------------------------------------------------------------
extern "C" void kernel_launch(void* const* d_in, const int* in_sizes, int n_in,
                              void* d_out, int out_size, void* d_ws, size_t ws_size,
                              hipStream_t stream) {
    const float* vgg   = (const float*)d_in[0];
    const float* tag   = (const float*)d_in[1];
    const int*   seqs  = (const int*)d_in[2];
    const float* emb   = (const float*)d_in[4];
    const float* Wimg  = (const float*)d_in[5];
    const float* bimg  = (const float*)d_in[6];
    const float* Wa    = (const float*)d_in[7];
    const float* Wb    = (const float*)d_in[8];
    const float* Wc    = (const float*)d_in[9];
    const float* Ua    = (const float*)d_in[10];
    const float* Ub    = (const float*)d_in[11];
    const float* Uc    = (const float*)d_in[12];
    const float* blstm = (const float*)d_in[13];
    const float* Wcls  = (const float*)d_in[14];
    const float* bcls  = (const float*)d_in[15];
    float* out = (float*)d_out;

    char* ws = (char*)d_ws;
    size_t off = 0;
    auto alloc = [&](size_t bytes) -> void* {
        void* p = ws + off;
        off = (off + bytes + 255) & ~(size_t)255;
        return p;
    };

    bf16* vgg_bf  = (bf16*)alloc((size_t)Bb * IMG * 2);
    bf16* tag_bf  = (bf16*)alloc((size_t)Bb * Ee * 2);
    bf16* Wimg_bf = (bf16*)alloc((size_t)384 * IMG * 2);        // N padded 300->384
    bf16* Wb_bf   = (bf16*)alloc((size_t)G4 * Ee * 2);
    bf16* Ub_bf   = (bf16*)alloc((size_t)G4 * Ee * 2);
    bf16* Wc_bf   = (bf16*)alloc((size_t)G4 * INP * 2);         // K padded 300->320
    bf16* Wa_bf   = (bf16*)alloc((size_t)G4 * Ff * 2);
    bf16* Ua_bf   = (bf16*)alloc((size_t)G4 * Ff * 2);
    bf16* Uc_bf   = (bf16*)alloc((size_t)G4 * Hh * 2);
    bf16* Wcls_bf = (bf16*)alloc((size_t)Vv * Hh * 2);
    bf16* x_bf    = (bf16*)alloc((size_t)TB * INP * 2);
    float* tagx   = (float*)alloc((size_t)Bb * G4 * 4);
    float* tagh   = (float*)alloc((size_t)Bb * G4 * 4);
    bf16* xf_bf   = (bf16*)alloc((size_t)TB * G4 * 2);
    float* xz     = (float*)alloc((size_t)TB * G4 * 4);
    bf16* h_bf    = (bf16*)alloc((size_t)Bb * Hh * 2);
    float* h_f    = (float*)alloc((size_t)Bb * Hh * 4);
    float* c_f    = (float*)alloc((size_t)Bb * Hh * 4);
    bf16* hf_bf   = (bf16*)alloc((size_t)Bb * G4 * 2);
    float* zbuf   = (float*)alloc((size_t)Bb * G4 * 4);
    bf16* hs_bf   = (bf16*)alloc((size_t)TB * Hh * 2);

    auto cvt = [&](bf16* d, const float* s, long long n) {
        k_cvt<<<dim3((unsigned)((n + 255) / 256)), dim3(256), 0, stream>>>(d, s, n);
    };
    auto gemm = [&](const bf16* A, int lda, long long a_z,
                    const bf16* Bm, int ldb, long long b_z,
                    int M, int N, int K, int Z,
                    float* oF, int ldoF, long long oFz,
                    bf16* oB, int ldoB, long long oBz,
                    const float* bias, long long bias_z,
                    const float* mul, int ldmul,
                    const float* add, int ldadd, long long add_z) {
        dim3 grid((unsigned)(M / 64), (unsigned)((N + 127) / 128), (unsigned)Z);
        k_gemm<<<grid, dim3(256), 0, stream>>>(A, lda, a_z, Bm, ldb, b_z, N, K,
                                               oF, ldoF, oFz, oB, ldoB, oBz,
                                               bias, bias_z, mul, ldmul,
                                               add, ldadd, add_z);
    };

    // ---- weight / activation conversions to bf16 -------------------------
    cvt(vgg_bf, vgg, (long long)Bb * IMG);
    cvt(tag_bf, tag, (long long)Bb * Ee);
    {
        long long n = (long long)384 * IMG;
        k_cvt_pad<<<dim3((unsigned)((n + 255) / 256)), dim3(256), 0, stream>>>(
            Wimg_bf, Wimg, 384, IMG, INw, IMG);
    }
    cvt(Wb_bf, Wb, (long long)G4 * Ee);
    cvt(Ub_bf, Ub, (long long)G4 * Ee);
    {
        long long n = (long long)G4 * INP;
        k_cvt_pad<<<dim3((unsigned)((n + 255) / 256)), dim3(256), 0, stream>>>(
            Wc_bf, Wc, G4, INP, G4, INw);
    }
    cvt(Wa_bf, Wa, (long long)G4 * Ff);
    cvt(Ua_bf, Ua, (long long)G4 * Ff);
    cvt(Uc_bf, Uc, (long long)G4 * Hh);
    cvt(Wcls_bf, Wcls, (long long)Vv * Hh);

    // ---- x: embedding gather (t>=1) + image projection (t==0) -----------
    k_build_x<<<dim3(TB), dim3(INP), 0, stream>>>(x_bf, emb, seqs);
    gemm(vgg_bf, IMG, 0, Wimg_bf, IMG, 0, Bb, INw, IMG, 1,
         nullptr, 0, 0, x_bf, INP, 0, bimg, 0, nullptr, 0, nullptr, 0, 0);

    // ---- tag factors ------------------------------------------------------
    gemm(tag_bf, Ee, 0, Wb_bf, Ee, 0, Bb, G4, Ee, 1,
         tagx, G4, 0, nullptr, 0, 0, nullptr, 0, nullptr, 0, nullptr, 0, 0);
    gemm(tag_bf, Ee, 0, Ub_bf, Ee, 0, Bb, G4, Ee, 1,
         tagh, G4, 0, nullptr, 0, 0, nullptr, 0, nullptr, 0, nullptr, 0, 0);

    // ---- xf = (x * Wc^T) . tagx  (bf16 out) ------------------------------
    gemm(x_bf, INP, 0, Wc_bf, INP, 0, TB, G4, INP, 1,
         nullptr, 0, 0, xf_bf, G4, 0, nullptr, 0, tagx, G4, nullptr, 0, 0);

    // ---- xz = per-gate xf_g * Wa_g^T + b_lstm  (f32 out, z-batched) ------
    gemm(xf_bf, G4, Ff, Wa_bf, Ff, (long long)Hh * Ff, TB, Hh, Ff, 4,
         xz, G4, Hh, nullptr, 0, 0, blstm, Hh, nullptr, 0, nullptr, 0, 0);

    // ---- recurrence -------------------------------------------------------
    k_zero_b<<<dim3((Bb * Hh) / 256), dim3(256), 0, stream>>>(h_bf, (long long)Bb * Hh);
    k_zero_f<<<dim3((Bb * Hh) / 256), dim3(256), 0, stream>>>(c_f, (long long)Bb * Hh);

    for (int t = 0; t < Tt; ++t) {
        // hf = (h * Uc^T) . tagh          [B, 4F] bf16
        gemm(h_bf, Hh, 0, Uc_bf, Hh, 0, Bb, G4, Hh, 1,
             nullptr, 0, 0, hf_bf, G4, 0, nullptr, 0, tagh, G4, nullptr, 0, 0);
        // z = xz[t] + per-gate hf_g * Ua_g^T   [B, 4H] f32
        gemm(hf_bf, G4, Ff, Ua_bf, Ff, (long long)Hh * Ff, Bb, Hh, Ff, 4,
             zbuf, G4, Hh, nullptr, 0, 0, nullptr, 0, nullptr, 0,
             xz + (long long)t * Bb * G4, G4, Hh);
        // gates + state update
        k_gates<<<dim3((Bb * Hh) / 256), dim3(256), 0, stream>>>(
            zbuf, c_f, h_f, h_bf, hs_bf, t);
    }

    // ---- logits = hs * Wcls^T + bcls -------------------------------------
    gemm(hs_bf, Hh, 0, Wcls_bf, Hh, 0, TB, Vv, Hh, 1,
         out, Vv, 0, nullptr, 0, 0, bcls, 0, nullptr, 0, nullptr, 0, 0);

    // ---- hT, cT tail ------------------------------------------------------
    k_tail<<<dim3((Bb * Hh) / 256), dim3(256), 0, stream>>>(
        out + (long long)TB * Vv, h_f, c_f);

    (void)in_sizes; (void)n_in; (void)out_size; (void)ws_size;
}
